// TriMul_47562467836629
// MI455X (gfx1250) — matrix operand order repaired and checked by reading, not compiled, not run on hardware
//
#include <hip/hip_runtime.h>

#define BB 2
#define SS 256
#define DIMD 128
#define HH 128
#define NROWS (BB*SS*SS)

typedef __attribute__((ext_vector_type(16))) __bf16 v16bf;
typedef __attribute__((ext_vector_type(8)))  float  v8f;
typedef __attribute__((ext_vector_type(2)))  float  v2f;

__device__ __forceinline__ v8f zero_v8f() {
    v8f v;
#pragma unroll
    for (int i = 0; i < 8; ++i) v[i] = 0.f;
    return v;
}

__device__ __forceinline__ float sigmoidf(float x) {
    return 1.0f / (1.0f + __expf(-x));
}

// ---------------------------------------------------------------------------
// Kernel A: convert the 5 projection weight matrices [H,D] f32 -> bf16 scratch
// ---------------------------------------------------------------------------
__global__ void wcvt_kernel(const float* __restrict__ wl, const float* __restrict__ wr,
                            const float* __restrict__ wlg, const float* __restrict__ wrg,
                            const float* __restrict__ wog, __bf16* __restrict__ wb) {
    int idx = blockIdx.x * 256 + threadIdx.x;
    if (idx >= 5 * DIMD * HH) return;
    int m = idx >> 14;         // which matrix (16384 elems each)
    int e = idx & 16383;
    const float* src = (m == 0) ? wl : (m == 1) ? wr : (m == 2) ? wlg : (m == 3) ? wrg : wog;
    wb[idx] = (__bf16)src[e];
}

// ---------------------------------------------------------------------------
// Kernel B: fused LayerNorm + 5 projections (bf16 WMMA) + mask/sigmoid gating
// Block = 128 threads (4 waves), processes 16 rows (= 16 consecutive k of one
// (b,i)) of [NROWS, D].  left/right stored TRANSPOSED as [B,H,S(i),S(k)] so
// the einsum kernel gets k-contiguous fragments; per-lane D-fragment values
// are k-consecutive, so each lane packs 8 bf16 into one 16B store.
// ---------------------------------------------------------------------------
__global__ __launch_bounds__(128) void proj_kernel(
    const float* __restrict__ x, const float* __restrict__ mask,
    const float* __restrict__ nw, const float* __restrict__ nb,
    const __bf16* __restrict__ wb,
    __bf16* __restrict__ leftT, __bf16* __restrict__ rightT, __bf16* __restrict__ ogp)
{
    __shared__ __bf16 xn[16][DIMD];
    __shared__ float rowmask[16];

    const int tid = threadIdx.x;
    const int r  = tid >> 3;      // row within tile 0..15
    const int c8 = tid & 7;       // 8 threads per row, 16 elems each
    const long row0 = (long)blockIdx.x * 16;
    const int bidx  = (int)(row0 >> 16);        // batch
    const int irow  = (int)((row0 >> 8) & 255); // i
    const int kbase = (int)(row0 & 255);        // k of first row in tile

    // ---- Phase 1: LayerNorm over D=128 ----
    const float* xrow = x + (row0 + r) * DIMD + c8 * 16;
    float v[16];
#pragma unroll
    for (int q = 0; q < 4; ++q) {
        float4 f = ((const float4*)xrow)[q];
        v[4*q+0] = f.x; v[4*q+1] = f.y; v[4*q+2] = f.z; v[4*q+3] = f.w;
    }
    float s = 0.f, s2 = 0.f;
#pragma unroll
    for (int q = 0; q < 16; ++q) { s += v[q]; s2 += v[q] * v[q]; }
#pragma unroll
    for (int off = 1; off < 8; off <<= 1) {
        s  += __shfl_xor(s,  off, 32);
        s2 += __shfl_xor(s2, off, 32);
    }
    const float mean = s * (1.0f / DIMD);
    const float var  = s2 * (1.0f / DIMD) - mean * mean;
    const float rstd = rsqrtf(var + 1e-5f);
#pragma unroll
    for (int q = 0; q < 16; ++q) {
        int d = c8 * 16 + q;
        float t = (v[q] - mean) * rstd * nw[d] + nb[d];
        xn[r][d] = (__bf16)t;
    }
    if (c8 == 0) rowmask[r] = mask[row0 + r];
    __syncthreads();

    // ---- Phase 2: 5 GEMMs [16xD] @ [DxH] via bf16 WMMA ----
    const int lane = tid & 31, w = tid >> 5;
    const int half = lane >> 4;
    const int mrow = lane & 15;
    const int n    = lane & 15;

    v8f acc[5][2];
#pragma unroll
    for (int q = 0; q < 5; ++q)
#pragma unroll
        for (int t = 0; t < 2; ++t) acc[q][t] = zero_v8f();

#pragma unroll
    for (int k0 = 0; k0 < 4; ++k0) {
        // A fragment 16x32 bf16: per lane two 16B-contiguous runs
        v16bf afrag;
#pragma unroll
        for (int p = 0; p < 8; ++p) {
            afrag[p]     = xn[mrow][k0 * 32 + half * 8 + p];
            afrag[8 + p] = xn[mrow][k0 * 32 + 16 + half * 8 + p];
        }
#pragma unroll
        for (int q = 0; q < 5; ++q) {
#pragma unroll
            for (int t = 0; t < 2; ++t) {
                int h0 = w * 32 + t * 16;
                const __bf16* Wrow = wb + q * DIMD * HH + (h0 + n) * DIMD + k0 * 32 + half * 8;
                v16bf bfrag;   // B(k,nn) = W[h0+nn][k]; per lane two 8-k runs (same order as A)
#pragma unroll
                for (int p = 0; p < 8; ++p) { bfrag[p] = Wrow[p]; bfrag[8 + p] = Wrow[16 + p]; }
                acc[q][t] = __builtin_amdgcn_wmma_f32_16x16x32_bf16(
                    false, afrag, false, bfrag, (short)0, acc[q][t], false, false);
            }
        }
    }

    // ---- Phase 3: gating epilogue ----
#pragma unroll
    for (int t = 0; t < 2; ++t) {
        int h0 = w * 32 + t * 16;
        int hcol = h0 + n;
        __align__(16) __bf16 ltmp[8];
        __align__(16) __bf16 rtmp[8];
#pragma unroll
        for (int e = 0; e < 8; ++e) {
            int m = e + half * 8;     // row in tile == k offset
            float mk = rowmask[m];
            float lv = acc[0][t][e] * mk * sigmoidf(acc[2][t][e]);
            float rv = acc[1][t][e] * mk * sigmoidf(acc[3][t][e]);
            float gv = sigmoidf(acc[4][t][e]);
            ltmp[e] = (__bf16)lv;
            rtmp[e] = (__bf16)rv;
            ogp[(row0 + m) * HH + hcol] = (__bf16)gv;   // [row,h] layout for kernel D
        }
        // transposed [b,h,i,k] store: 8 k-consecutive bf16 -> one 16B store
        size_t o = ((size_t)(bidx * HH + hcol) * SS + irow) * SS + kbase + half * 8;
        *(uint4*)(leftT + o)  = *(const uint4*)ltmp;
        *(uint4*)(rightT + o) = *(const uint4*)rtmp;
    }
}

// ---------------------------------------------------------------------------
// Kernel C: tri[b,i,j,h] = sum_k left[b,i,k,h] * right[b,j,k,h]
// left/right in [B,H,S,S] layout -> every WMMA fragment is k-contiguous in
// global memory: no LDS, 2 global b128 loads per fragment, served from L2.
// Block = 256 threads (8 waves = 8 h); each wave computes a 2x2 grid of
// 16x16 (i,j) tiles for its h -> 2 b128 loads per WMMA.
// ---------------------------------------------------------------------------
__global__ __launch_bounds__(256) void trimul_kernel(
    const __bf16* __restrict__ Lh, const __bf16* __restrict__ Rh,
    __bf16* __restrict__ trip)
{
    const int tid = threadIdx.x;
    const int lane = tid & 31, w = tid >> 5;
    const int bi = blockIdx.x;          // 2048 blocks
    const int jt = bi & 7;              // j 32-tile
    const int it = (bi >> 3) & 7;       // i 32-tile
    const int hg = (bi >> 6) & 15;      // h group of 8
    const int b  = bi >> 10;            // batch
    const int h  = hg * 8 + w;
    const int i0 = it * 32, j0 = jt * 32;
    const int half = lane >> 4;
    const int mn   = lane & 15;

    const __bf16* Abase = Lh + (size_t)(b * HH + h) * SS * SS;
    const __bf16* Bbase = Rh + (size_t)(b * HH + h) * SS * SS;

    v8f acc[2][2];
#pragma unroll
    for (int ti = 0; ti < 2; ++ti)
#pragma unroll
        for (int tj = 0; tj < 2; ++tj) acc[ti][tj] = zero_v8f();

    for (int kc = 0; kc < 8; ++kc) {
        const int k0 = kc * 32;
        v16bf af[2], bf[2];
#pragma unroll
        for (int ti = 0; ti < 2; ++ti) {
            const __bf16* arow = Abase + (size_t)(i0 + ti * 16 + mn) * SS + k0;
#pragma unroll
            for (int p = 0; p < 8; ++p) {
                af[ti][p]     = arow[half * 8 + p];
                af[ti][8 + p] = arow[16 + half * 8 + p];
            }
            if (kc < 7) __builtin_prefetch((const void*)(arow + 32), 0, 1);
        }
#pragma unroll
        for (int tj = 0; tj < 2; ++tj) {
            const __bf16* brow = Bbase + (size_t)(j0 + tj * 16 + mn) * SS + k0;
#pragma unroll
            for (int p = 0; p < 8; ++p) { bf[tj][p] = brow[half * 8 + p]; bf[tj][8 + p] = brow[16 + half * 8 + p]; }
            if (kc < 7) __builtin_prefetch((const void*)(brow + 32), 0, 1);
        }
#pragma unroll
        for (int ti = 0; ti < 2; ++ti)
#pragma unroll
            for (int tj = 0; tj < 2; ++tj)
                acc[ti][tj] = __builtin_amdgcn_wmma_f32_16x16x32_bf16(
                    false, af[ti], false, bf[tj], (short)0, acc[ti][tj], false, false);
    }

    // Store bf16 (matches reference's f32->bf16 round-trip of the einsum)
#pragma unroll
    for (int ti = 0; ti < 2; ++ti)
#pragma unroll
        for (int tj = 0; tj < 2; ++tj)
#pragma unroll
            for (int e = 0; e < 8; ++e) {
                int i = i0 + ti * 16 + e + half * 8;
                int j = j0 + tj * 16 + mn;
                trip[((size_t)((b * SS + i) * SS + j)) * HH + h] = (__bf16)acc[ti][tj][e];
            }
}

// ---------------------------------------------------------------------------
// Kernel D: fused LN over H * out_gate, then f32 WMMA output projection
// Block = 128 threads (4 waves), 16 rows.
// ---------------------------------------------------------------------------
__global__ __launch_bounds__(128) void outproj_kernel(
    const __bf16* __restrict__ trip, const __bf16* __restrict__ ogp,
    const float* __restrict__ onw, const float* __restrict__ onb,
    const float* __restrict__ wout, float* __restrict__ outp)
{
    __shared__ float gs[16][HH];

    const int tid = threadIdx.x;
    const int r  = tid >> 3;
    const int c8 = tid & 7;
    const long row0 = (long)blockIdx.x * 16;

    // ---- Phase 1: LN over H, multiply by out_gate ----
    const __bf16* trow = trip + (row0 + r) * HH + c8 * 16;
    const __bf16* grow = ogp  + (row0 + r) * HH + c8 * 16;
    float v[16];
#pragma unroll
    for (int q = 0; q < 16; ++q) v[q] = (float)trow[q];
    float s = 0.f, s2 = 0.f;
#pragma unroll
    for (int q = 0; q < 16; ++q) { s += v[q]; s2 += v[q] * v[q]; }
#pragma unroll
    for (int off = 1; off < 8; off <<= 1) {
        s  += __shfl_xor(s,  off, 32);
        s2 += __shfl_xor(s2, off, 32);
    }
    const float mean = s * (1.0f / HH);
    const float var  = s2 * (1.0f / HH) - mean * mean;
    const float rstd = rsqrtf(var + 1e-5f);
#pragma unroll
    for (int q = 0; q < 16; ++q) {
        int h = c8 * 16 + q;
        float t = (v[q] - mean) * rstd * onw[h] + onb[h];
        gs[r][h] = t * (float)grow[q];
    }
    __syncthreads();

    // ---- Phase 2: [16xH] @ [HxD] via f32 WMMA 16x16x4 ----
    const int lane = tid & 31, w = tid >> 5;
    const int half = lane >> 4;
    const int mrow = lane & 15;
    const int n    = lane & 15;

    v8f acc[2];
    acc[0] = zero_v8f();
    acc[1] = zero_v8f();

#pragma unroll
    for (int kk = 0; kk < 32; ++kk) {
        const int kb = kk * 4;
        v2f a;  // A(m,k): VGPR0=K(2*half), VGPR1=K(2*half+1)
        {
            float2 f = *(const float2*)(&gs[mrow][kb + 2 * half]);
            a.x = f.x; a.y = f.y;
        }
#pragma unroll
        for (int t = 0; t < 2; ++t) {
            int n0 = w * 32 + t * 16;
            v2f bb; // B(k,n) = w_out[n0+n][k]
            {
                float2 f = *(const float2*)(wout + (n0 + n) * HH + kb + 2 * half);
                bb.x = f.x; bb.y = f.y;
            }
            acc[t] = __builtin_amdgcn_wmma_f32_16x16x4_f32(
                false, a, false, bb, (short)0, acc[t], false, false);
        }
    }

#pragma unroll
    for (int t = 0; t < 2; ++t) {
        int n0 = w * 32 + t * 16;
#pragma unroll
        for (int e = 0; e < 8; ++e) {
            int mm = e + half * 8;
            outp[(row0 + mm) * DIMD + n0 + n] = acc[t][e];
        }
    }
}

// ---------------------------------------------------------------------------
extern "C" void kernel_launch(void* const* d_in, const int* in_sizes, int n_in,
                              void* d_out, int out_size, void* d_ws, size_t ws_size,
                              hipStream_t stream) {
    const float* x    = (const float*)d_in[0];
    const float* mask = (const float*)d_in[1];
    const float* nw   = (const float*)d_in[2];
    const float* nb   = (const float*)d_in[3];
    const float* wl   = (const float*)d_in[4];
    const float* wr   = (const float*)d_in[5];
    const float* wlg  = (const float*)d_in[6];
    const float* wrg  = (const float*)d_in[7];
    const float* wog  = (const float*)d_in[8];
    const float* onw  = (const float*)d_in[9];
    const float* onb  = (const float*)d_in[10];
    const float* wout = (const float*)d_in[11];
    float* out = (float*)d_out;

    char* ws = (char*)d_ws;
    size_t off = 0;
    __bf16* wb     = (__bf16*)(ws + off); off += (size_t)5 * DIMD * HH * 2;
    off = (off + 255) & ~(size_t)255;
    __bf16* leftT  = (__bf16*)(ws + off); off += (size_t)NROWS * HH * 2;  // [B,H,S,S]
    __bf16* rightT = (__bf16*)(ws + off); off += (size_t)NROWS * HH * 2;  // [B,H,S,S]
    __bf16* ogp    = (__bf16*)(ws + off); off += (size_t)NROWS * HH * 2;  // [B,S,S,H]
    __bf16* trip   = (__bf16*)(ws + off); off += (size_t)NROWS * HH * 2;  // [B,S,S,H]

    wcvt_kernel<<<(5 * DIMD * HH + 255) / 256, 256, 0, stream>>>(wl, wr, wlg, wrg, wog, wb);
    proj_kernel<<<NROWS / 16, 128, 0, stream>>>(x, mask, nw, nb, wb, leftT, rightT, ogp);
    trimul_kernel<<<BB * 16 * (SS / 32) * (SS / 32), 256, 0, stream>>>(leftT, rightT, trip);
    outproj_kernel<<<NROWS / 16, 128, 0, stream>>>(trip, ogp, onw, onb, wout, out);
}
